// TopBlock_63350767616772
// MI455X (gfx1250) — compile-verified, matrix operand order checked by reading, not run on hardware
//
#include <hip/hip_runtime.h>
#include <hip/hip_bf16.h>
#include <math.h>

#define TOK 2048
#define DIM 1024
#define HEADS 8
#define QK_HD 128
#define V_HD 128
#define NOPE 64
#define ROPE 64
#define KV_LORA 256
#define N_EXP 32
#define TOPK 4
#define INTER 512
#define EPS 1e-6f

typedef unsigned short u16;
typedef unsigned int u32;
typedef __bf16 bf16x16 __attribute__((ext_vector_type(16)));
typedef float f32x8 __attribute__((ext_vector_type(8)));

// ---- CDNA5 async global->LDS copy (ASYNCcnt path), guarded for portability --
#if defined(__has_builtin)
#if __has_builtin(__builtin_amdgcn_global_load_async_to_lds_b128) && \
    __has_builtin(__builtin_amdgcn_s_wait_asynccnt)
#define HAVE_ASYNC_LDS 1
#endif
#endif

typedef int vi4 __attribute__((vector_size(16)));
typedef __attribute__((address_space(1))) vi4* gvi4p;
typedef __attribute__((address_space(3))) vi4* lvi4p;

__device__ __forceinline__ void copy16_to_lds(const u16* g, u16* l) {
#ifdef HAVE_ASYNC_LDS
  __builtin_amdgcn_global_load_async_to_lds_b128((gvi4p)g, (lvi4p)l, 0, 0);
#else
  *(uint4*)l = *(const uint4*)g;
#endif
}
__device__ __forceinline__ void async_copy_wait() {
#ifdef HAVE_ASYNC_LDS
  __builtin_amdgcn_s_wait_asynccnt(0);
#endif
}

__device__ __forceinline__ u16 f2bf(float f) {
  u32 u = __float_as_uint(f);
  return (u16)((u + 0x7FFFu + ((u >> 16) & 1u)) >> 16);
}

__device__ __forceinline__ f32x8 wmma_bf16(bf16x16 a, bf16x16 b, f32x8 c) {
  return __builtin_amdgcn_wmma_f32_16x16x32_bf16(false, a, false, b, (short)0, c,
                                                 false, false);
}

// Load one 16x32 bf16 fragment (A-layout; also B-layout from a transposed tile)
// from a row-major [16 x rowstride] region of bf16 (u16 bits).
// Per ISA: lanes 0-15 hold row m=lane, K=0..7 & 16..23; lanes 16-31 hold
// row m=lane-16, K=8..15 & 24..31.  -> two 16B contiguous loads per lane.
__device__ __forceinline__ bf16x16 load_frag(const u16* base, int rowstride) {
  int lane = threadIdx.x & 31;
  const u16* p0 = base + (lane & 15) * rowstride + ((lane < 16) ? 0 : 8);
  union { uint4 q[2]; bf16x16 v; } u;
  u.q[0] = *(const uint4*)p0;
  u.q[1] = *(const uint4*)(p0 + 16);
  return u.v;
}

__device__ __forceinline__ float block_reduce_sum256(float v, float* smem) {
  int tid = threadIdx.x;
  smem[tid] = v;
  __syncthreads();
  for (int s = 128; s > 0; s >>= 1) {
    if (tid < s) smem[tid] += smem[tid + s];
    __syncthreads();
  }
  float r = smem[0];
  __syncthreads();
  return r;
}

// ---------------- rmsnorm: one token per block (dim 1024) ----------------
__global__ void __launch_bounds__(256) rmsnorm_kernel(
    const float* __restrict__ x, const float* __restrict__ w,
    u16* __restrict__ obf, float* __restrict__ of) {
  __shared__ float red[256];
  int t = blockIdx.x, tid = threadIdx.x;
  float4 a = ((const float4*)(x + (size_t)t * DIM))[tid];
  float tot = block_reduce_sum256(a.x * a.x + a.y * a.y + a.z * a.z + a.w * a.w, red);
  float inv = rsqrtf(tot * (1.0f / DIM) + EPS);
  float4 wv = ((const float4*)w)[tid];
  float r0 = a.x * inv * wv.x, r1 = a.y * inv * wv.y;
  float r2 = a.z * inv * wv.z, r3 = a.w * inv * wv.w;
  size_t o = (size_t)t * DIM + tid * 4;
  obf[o + 0] = f2bf(r0); obf[o + 1] = f2bf(r1);
  obf[o + 2] = f2bf(r2); obf[o + 3] = f2bf(r3);
  if (of) { of[o + 0] = r0; of[o + 1] = r1; of[o + 2] = r2; of[o + 3] = r3; }
}

// ---------------- kv-lora rmsnorm (dim 256) ----------------
__global__ void __launch_bounds__(256) kvnorm_kernel(
    const float* __restrict__ kv, const float* __restrict__ w, u16* __restrict__ cbf) {
  __shared__ float red[256];
  int t = blockIdx.x, tid = threadIdx.x;
  float v = kv[(size_t)t * (KV_LORA + ROPE) + tid];
  float tot = block_reduce_sum256(v * v, red);
  float inv = rsqrtf(tot * (1.0f / KV_LORA) + EPS);
  cbf[(size_t)t * KV_LORA + tid] = f2bf(v * inv * w[tid]);
}

// ---------------- generic WMMA GEMM: C[M,N] = A_bf16[M,K] @ W_f32[K,N] ----
// Block tile 128x64, BK=32, 256 threads (8 waves, each 32x32).
// EPI 0: store f32;  EPI 1: store bf16 * scale;  EPI 2: store f32 + res.
template <int EPI>
__global__ void __launch_bounds__(256, 1) gemm_kernel(
    const u16* __restrict__ A, const float* __restrict__ W,
    float* __restrict__ Cf, u16* __restrict__ Cb,
    const float* __restrict__ res, int N, int K, float scale) {
  __shared__ u16 As[128 * 32];
  __shared__ u16 Bs[64 * 32];  // transposed [n][k]
  int tid = threadIdx.x, wave = tid >> 5, lane = tid & 31;
  int m0 = blockIdx.y * 128, n0 = blockIdx.x * 64;
  int wm = (wave & 3) * 32, wn = (wave >> 2) * 32;
  f32x8 acc[2][2];
  for (int a = 0; a < 2; a++)
    for (int b = 0; b < 2; b++)
      for (int i = 0; i < 8; i++) acc[a][b][i] = 0.f;
  int arow = tid >> 1, acol = (tid & 1) * 16;
  int bk = tid >> 3, bn = (tid & 7) * 8;
  for (int k0 = 0; k0 < K; k0 += 32) {
    const u16* ga = A + (size_t)(m0 + arow) * K + k0 + acol;
    copy16_to_lds(ga, &As[arow * 32 + acol]);
    copy16_to_lds(ga + 8, &As[arow * 32 + acol + 8]);
    const float* gw = W + (size_t)(k0 + bk) * N + n0 + bn;
    float4 w0 = ((const float4*)gw)[0];
    float4 w1 = ((const float4*)gw)[1];
    Bs[(bn + 0) * 32 + bk] = f2bf(w0.x); Bs[(bn + 1) * 32 + bk] = f2bf(w0.y);
    Bs[(bn + 2) * 32 + bk] = f2bf(w0.z); Bs[(bn + 3) * 32 + bk] = f2bf(w0.w);
    Bs[(bn + 4) * 32 + bk] = f2bf(w1.x); Bs[(bn + 5) * 32 + bk] = f2bf(w1.y);
    Bs[(bn + 6) * 32 + bk] = f2bf(w1.z); Bs[(bn + 7) * 32 + bk] = f2bf(w1.w);
    async_copy_wait();
    __syncthreads();
    bf16x16 a0 = load_frag(&As[(wm + 0) * 32], 32);
    bf16x16 a1 = load_frag(&As[(wm + 16) * 32], 32);
    bf16x16 b0 = load_frag(&Bs[(wn + 0) * 32], 32);
    bf16x16 b1 = load_frag(&Bs[(wn + 16) * 32], 32);
    acc[0][0] = wmma_bf16(a0, b0, acc[0][0]);
    acc[0][1] = wmma_bf16(a0, b1, acc[0][1]);
    acc[1][0] = wmma_bf16(a1, b0, acc[1][0]);
    acc[1][1] = wmma_bf16(a1, b1, acc[1][1]);
    __syncthreads();
  }
  int col = lane & 15, rh = (lane >> 4) * 8;
  for (int mi = 0; mi < 2; mi++)
    for (int ni = 0; ni < 2; ni++)
      for (int i = 0; i < 8; i++) {
        int r = m0 + wm + mi * 16 + i + rh;
        int c = n0 + wn + ni * 16 + col;
        size_t idx = (size_t)r * N + c;
        float v = acc[mi][ni][i];
        if (EPI == 0) Cf[idx] = v;
        else if (EPI == 1) Cb[idx] = f2bf(v * scale);
        else Cf[idx] = v + res[idx];
      }
}

// ---------------- pack K/V into [b*h][s][128] bf16 ----------------
__global__ void __launch_bounds__(256) pack_kv_kernel(
    const u16* __restrict__ kvp, const float* __restrict__ kv,
    u16* __restrict__ Kb, u16* __restrict__ Vb) {
  int t = blockIdx.x;
  int b = t >> 10, s = t & 1023;
  for (int j = threadIdx.x; j < HEADS * 128; j += 256) {
    int h = j >> 7, d = j & 127;
    size_t o = (((size_t)(b * HEADS + h)) * 1024 + s) * 128 + d;
    u16 kval = (d < NOPE)
                   ? kvp[(size_t)t * 1536 + h * (NOPE + V_HD) + d]
                   : f2bf(kv[(size_t)t * (KV_LORA + ROPE) + KV_LORA + (d - NOPE)]);
    Kb[o] = kval;
    Vb[o] = kvp[(size_t)t * 1536 + h * (NOPE + V_HD) + NOPE + d];
  }
}

// ---------------- flash attention: 128 q-rows per block, 16 per wave --------
__global__ void __launch_bounds__(256, 1) attn_kernel(
    const u16* __restrict__ Q, const u16* __restrict__ Kb,
    const u16* __restrict__ Vb, u16* __restrict__ O) {
  __shared__ u16 Ks[32 * 128];      // [key][d]
  __shared__ u16 Vt[128 * 32];      // [d][key] (transposed)
  __shared__ u16 Ps[8][16 * 32];    // per-wave P staging
  int bh = blockIdx.y;
  int b = bh >> 3, h = bh & 7;
  int q0 = blockIdx.x * 128;
  int tid = threadIdx.x, wave = tid >> 5, lane = tid & 31;
  int qw = q0 + wave * 16;
  int col = lane & 15, rh = (lane >> 4) * 8;
  bf16x16 qf[4];
  const u16* qbase = Q + ((size_t)(b * 1024 + qw)) * DIM + h * QK_HD;
  for (int kk = 0; kk < 4; kk++) qf[kk] = load_frag(qbase + kk * 32, DIM);
  f32x8 o[8];
  float mx[8], l[8];
  for (int n = 0; n < 8; n++)
    for (int i = 0; i < 8; i++) o[n][i] = 0.f;
  for (int i = 0; i < 8; i++) { mx[i] = -__builtin_inff(); l[i] = 0.f; }
  int nsteps = (q0 + 128) >> 5;
  int kr = tid >> 3, kc = (tid & 7) * 16;
  const u16* kvbase = Kb + ((size_t)bh * 1024 + kr) * 128 + kc;
  const u16* vvbase = Vb + ((size_t)bh * 1024 + kr) * 128 + kc;
  for (int st = 0; st < nsteps; st++) {
    int s0 = st * 32;
    // K tile: raw bf16 copy -> async to LDS
    copy16_to_lds(kvbase + (size_t)s0 * 128, &Ks[kr * 128 + kc]);
    copy16_to_lds(kvbase + (size_t)s0 * 128 + 8, &Ks[kr * 128 + kc + 8]);
    // V tile: transpose on the way in (VALU path)
    const uint4* gvv = (const uint4*)(vvbase + (size_t)s0 * 128);
    uint4 v0v = gvv[0], v1v = gvv[1];
    u16 tmp[16];
    *(uint4*)&tmp[0] = v0v;
    *(uint4*)&tmp[8] = v1v;
    for (int j = 0; j < 16; j++) Vt[(kc + j) * 32 + kr] = tmp[j];
    async_copy_wait();
    __syncthreads();
    f32x8 sfr[2];
    for (int tt = 0; tt < 2; tt++)
      for (int i = 0; i < 8; i++) sfr[tt][i] = 0.f;
    for (int kk = 0; kk < 4; kk++) {
      bf16x16 kb0 = load_frag(&Ks[0 * 128 + kk * 32], 128);
      bf16x16 kb1 = load_frag(&Ks[16 * 128 + kk * 32], 128);
      sfr[0] = wmma_bf16(qf[kk], kb0, sfr[0]);
      sfr[1] = wmma_bf16(qf[kk], kb1, sfr[1]);
    }
    if (s0 + 31 > qw) {  // causal mask (also fully masks tiles above the row band)
      for (int tt = 0; tt < 2; tt++)
        for (int i = 0; i < 8; i++) {
          int key = s0 + tt * 16 + col;
          int row = qw + i + rh;
          if (key > row) sfr[tt][i] = -__builtin_inff();
        }
    }
    for (int i = 0; i < 8; i++) {
      float vmax = fmaxf(sfr[0][i], sfr[1][i]);
      for (int d2 = 1; d2 < 16; d2 <<= 1) vmax = fmaxf(vmax, __shfl_xor(vmax, d2, 32));
      float mnew = fmaxf(mx[i], vmax);
      float scl = __expf(mx[i] - mnew);
      mx[i] = mnew;
      float p0 = __expf(sfr[0][i] - mnew);
      float p1 = __expf(sfr[1][i] - mnew);
      sfr[0][i] = p0; sfr[1][i] = p1;
      float rs = p0 + p1;
      for (int d2 = 1; d2 < 16; d2 <<= 1) rs += __shfl_xor(rs, d2, 32);
      l[i] = l[i] * scl + rs;
      for (int n = 0; n < 8; n++) o[n][i] *= scl;
    }
    u16* pst = &Ps[wave][0];
    for (int tt = 0; tt < 2; tt++)
      for (int i = 0; i < 8; i++) pst[(i + rh) * 32 + tt * 16 + col] = f2bf(sfr[tt][i]);
    bf16x16 pf = load_frag(pst, 32);
    for (int n = 0; n < 8; n++) {
      bf16x16 vf = load_frag(&Vt[(n * 16) * 32], 32);
      o[n] = wmma_bf16(pf, vf, o[n]);
    }
    __syncthreads();
  }
  for (int i = 0; i < 8; i++) l[i] = 1.0f / l[i];
  for (int n = 0; n < 8; n++)
    for (int i = 0; i < 8; i++) {
      int row = qw + i + rh;
      int d = n * 16 + col;
      O[((size_t)(b * 1024 + row)) * DIM + h * V_HD + d] = f2bf(o[n][i] * l[i]);
    }
}

// ---------------- gating (one wave per token) ----------------
__global__ void __launch_bounds__(32) gate_kernel(
    const float* __restrict__ xf, const float* __restrict__ gw,
    const float* __restrict__ gb, int* __restrict__ ti,
    float* __restrict__ tw, int* __restrict__ counts) {
  __shared__ float sc[N_EXP];
  int t = blockIdx.x, lane = threadIdx.x;
  const float4* xp = (const float4*)(xf + (size_t)t * DIM);
  const float4* wp = (const float4*)(gw + (size_t)lane * DIM);
  float acc = 0.f;
  for (int k = 0; k < DIM / 4; k++) {
    float4 a = xp[k], b = wp[k];
    acc += a.x * b.x + a.y * b.y + a.z * b.z + a.w * b.w;
  }
  sc[lane] = 1.0f / (1.0f + __expf(-acc));
  __syncthreads();
  if (lane == 0) {
    const float NINF = -__builtin_inff();
    float orig[N_EXP], s[N_EXP];
    for (int e = 0; e < N_EXP; e++) { orig[e] = sc[e]; s[e] = sc[e] + gb[e]; }
    float gsc[8];
    for (int g = 0; g < 8; g++) {
      float m1 = NINF, m2 = NINF;
      for (int j = 0; j < 4; j++) {
        float v = s[4 * g + j];
        if (v > m1) { m2 = m1; m1 = v; } else if (v > m2) m2 = v;
      }
      gsc[g] = m1 + m2;
    }
    bool keep[8];
    for (int g = 0; g < 8; g++) keep[g] = false;
    for (int r = 0; r < 4; r++) {
      int bi = 0; float bv = NINF;
      for (int g = 0; g < 8; g++)
        if (!keep[g] && gsc[g] > bv) { bv = gsc[g]; bi = g; }
      keep[bi] = true;
    }
    for (int e = 0; e < N_EXP; e++)
      if (!keep[e >> 2]) s[e] = NINF;
    bool used[N_EXP];
    for (int e = 0; e < N_EXP; e++) used[e] = false;
    int idx[TOPK]; float w[TOPK]; float wsum = 0.f;
    for (int r = 0; r < TOPK; r++) {
      int bi = 0; float bv = NINF;
      for (int e = 0; e < N_EXP; e++)
        if (!used[e] && s[e] > bv) { bv = s[e]; bi = e; }
      used[bi] = true;
      idx[r] = bi; w[r] = orig[bi]; wsum += w[r];
    }
    float inv = 1.0f / (wsum + 1e-20f);
    for (int r = 0; r < TOPK; r++) {
      ti[t * TOPK + r] = idx[r];
      tw[t * TOPK + r] = w[r] * inv;  // ROUTE_SCALE == 1
      atomicAdd(&counts[idx[r]], 1);
    }
  }
}

__global__ void zero32_kernel(int* p) {
  if (threadIdx.x < N_EXP) p[threadIdx.x] = 0;
}

__global__ void offsets_kernel(const int* __restrict__ counts, int* __restrict__ offs,
                               int* __restrict__ fill) {
  if (threadIdx.x == 0) {
    int run = 0;
    for (int e = 0; e < N_EXP; e++) { offs[e] = run; run += counts[e]; }
  }
  if (threadIdx.x < N_EXP) fill[threadIdx.x] = 0;
}

__global__ void scatter_kernel(const int* __restrict__ ti, const float* __restrict__ tw,
                               const int* __restrict__ offs, int* __restrict__ fill,
                               int* __restrict__ perm, float* __restrict__ pw) {
  int t = blockIdx.x, r = threadIdx.x;
  if (r < TOPK) {
    int e = ti[t * TOPK + r];
    int slot = atomicAdd(&fill[e], 1);
    perm[offs[e] + slot] = t;
    pw[offs[e] + slot] = tw[t * TOPK + r];
  }
}

// ---------------- MoE gate/up fused GEMM: H = silu(X@W1) * (X@W3) ----------
// 64x64 block tile; gathered rows via perm (or identity when counts==nullptr).
__global__ void __launch_bounds__(256, 1) moe_gateup_kernel(
    const u16* __restrict__ X, const float* __restrict__ W1,
    const float* __restrict__ W3, long estride,
    const int* __restrict__ perm, const int* __restrict__ counts,
    const int* __restrict__ offs, int Mtotal, u16* __restrict__ H) {
  __shared__ int rows[64];
  __shared__ u16 As[64 * 32];
  __shared__ u16 B1s[64 * 32];
  __shared__ u16 B3s[64 * 32];
  int e = blockIdx.z;
  int cnt = counts ? counts[e] : Mtotal;
  int off = counts ? offs[e] : 0;
  int m0 = blockIdx.y * 64;
  if (m0 >= cnt) return;
  int n0 = blockIdx.x * 64;
  const float* w1 = W1 + (size_t)e * estride;
  const float* w3 = W3 + (size_t)e * estride;
  int tid = threadIdx.x, wave = tid >> 5, lane = tid & 31;
  if (tid < 64) {
    int r = m0 + tid;
    int rr = (r < cnt) ? r : (cnt - 1);
    rows[tid] = counts ? perm[off + rr] : rr;
  }
  __syncthreads();
  int wm = (wave & 3) * 16, wn = (wave >> 2) * 32;
  f32x8 a1[2], a3[2];
  for (int ni = 0; ni < 2; ni++)
    for (int i = 0; i < 8; i++) { a1[ni][i] = 0.f; a3[ni][i] = 0.f; }
  int ar = tid >> 2, ac = (tid & 3) * 8;
  int bk = tid >> 3, bn = (tid & 7) * 8;
  for (int k0 = 0; k0 < DIM; k0 += 32) {
    copy16_to_lds(X + (size_t)rows[ar] * DIM + k0 + ac, &As[ar * 32 + ac]);
    const float* g1 = w1 + (size_t)(k0 + bk) * INTER + n0 + bn;
    const float* g3 = w3 + (size_t)(k0 + bk) * INTER + n0 + bn;
    float4 p0 = ((const float4*)g1)[0], p1 = ((const float4*)g1)[1];
    float4 q0 = ((const float4*)g3)[0], q1 = ((const float4*)g3)[1];
    B1s[(bn + 0) * 32 + bk] = f2bf(p0.x); B1s[(bn + 1) * 32 + bk] = f2bf(p0.y);
    B1s[(bn + 2) * 32 + bk] = f2bf(p0.z); B1s[(bn + 3) * 32 + bk] = f2bf(p0.w);
    B1s[(bn + 4) * 32 + bk] = f2bf(p1.x); B1s[(bn + 5) * 32 + bk] = f2bf(p1.y);
    B1s[(bn + 6) * 32 + bk] = f2bf(p1.z); B1s[(bn + 7) * 32 + bk] = f2bf(p1.w);
    B3s[(bn + 0) * 32 + bk] = f2bf(q0.x); B3s[(bn + 1) * 32 + bk] = f2bf(q0.y);
    B3s[(bn + 2) * 32 + bk] = f2bf(q0.z); B3s[(bn + 3) * 32 + bk] = f2bf(q0.w);
    B3s[(bn + 4) * 32 + bk] = f2bf(q1.x); B3s[(bn + 5) * 32 + bk] = f2bf(q1.y);
    B3s[(bn + 6) * 32 + bk] = f2bf(q1.z); B3s[(bn + 7) * 32 + bk] = f2bf(q1.w);
    async_copy_wait();
    __syncthreads();
    bf16x16 af = load_frag(&As[wm * 32], 32);
    for (int ni = 0; ni < 2; ni++) {
      bf16x16 b1f = load_frag(&B1s[(wn + ni * 16) * 32], 32);
      bf16x16 b3f = load_frag(&B3s[(wn + ni * 16) * 32], 32);
      a1[ni] = wmma_bf16(af, b1f, a1[ni]);
      a3[ni] = wmma_bf16(af, b3f, a3[ni]);
    }
    __syncthreads();
  }
  int col = lane & 15, rh = (lane >> 4) * 8;
  for (int ni = 0; ni < 2; ni++)
    for (int i = 0; i < 8; i++) {
      int rl = wm + i + rh;
      if (m0 + rl < cnt) {
        float g = a1[ni][i], u = a3[ni][i];
        float hv = g * (1.0f / (1.0f + __expf(-g))) * u;
        H[(size_t)(off + m0 + rl) * INTER + n0 + wn + ni * 16 + col] = f2bf(hv);
      }
    }
}

// ---------------- MoE down GEMM + weighted atomic scatter-add --------------
__global__ void __launch_bounds__(256, 1) moe_down_kernel(
    const u16* __restrict__ H, const float* __restrict__ W2,
    long estride, const int* __restrict__ perm,
    const float* __restrict__ pw, const int* __restrict__ counts,
    const int* __restrict__ offs, int Mtotal, float* __restrict__ Out) {
  __shared__ int toks[64];
  __shared__ float wts[64];
  __shared__ u16 As[64 * 32];
  __shared__ u16 Bs[64 * 32];
  int e = blockIdx.z;
  int cnt = counts ? counts[e] : Mtotal;
  int off = counts ? offs[e] : 0;
  int m0 = blockIdx.y * 64;
  if (m0 >= cnt) return;
  int n0 = blockIdx.x * 64;
  const float* w2 = W2 + (size_t)e * estride;
  int tid = threadIdx.x, wave = tid >> 5, lane = tid & 31;
  if (tid < 64) {
    int r = m0 + tid;
    int rr = (r < cnt) ? r : (cnt - 1);
    toks[tid] = counts ? perm[off + rr] : rr;
    wts[tid] = counts ? pw[off + rr] : 1.0f;
  }
  __syncthreads();
  int wm = (wave & 3) * 16, wn = (wave >> 2) * 32;
  f32x8 acc[2];
  for (int ni = 0; ni < 2; ni++)
    for (int i = 0; i < 8; i++) acc[ni][i] = 0.f;
  int ar = tid >> 2, ac = (tid & 3) * 8;
  int bk = tid >> 3, bn = (tid & 7) * 8;
  {
    int r = m0 + ar;
    int rr = (r < cnt) ? r : (cnt - 1);
    size_t hrow = counts ? (size_t)(off + rr) : (size_t)rr;
    const u16* abase = H + hrow * INTER + ac;
    for (int k0 = 0; k0 < INTER; k0 += 32) {
      copy16_to_lds(abase + k0, &As[ar * 32 + ac]);
      const float* gw = w2 + (size_t)(k0 + bk) * DIM + n0 + bn;
      float4 p0 = ((const float4*)gw)[0], p1 = ((const float4*)gw)[1];
      Bs[(bn + 0) * 32 + bk] = f2bf(p0.x); Bs[(bn + 1) * 32 + bk] = f2bf(p0.y);
      Bs[(bn + 2) * 32 + bk] = f2bf(p0.z); Bs[(bn + 3) * 32 + bk] = f2bf(p0.w);
      Bs[(bn + 4) * 32 + bk] = f2bf(p1.x); Bs[(bn + 5) * 32 + bk] = f2bf(p1.y);
      Bs[(bn + 6) * 32 + bk] = f2bf(p1.z); Bs[(bn + 7) * 32 + bk] = f2bf(p1.w);
      async_copy_wait();
      __syncthreads();
      bf16x16 af = load_frag(&As[wm * 32], 32);
      for (int ni = 0; ni < 2; ni++) {
        bf16x16 bf = load_frag(&Bs[(wn + ni * 16) * 32], 32);
        acc[ni] = wmma_bf16(af, bf, acc[ni]);
      }
      __syncthreads();
    }
  }
  int col = lane & 15, rh = (lane >> 4) * 8;
  for (int ni = 0; ni < 2; ni++)
    for (int i = 0; i < 8; i++) {
      int rl = wm + i + rh;
      if (m0 + rl < cnt) {
        float v = acc[ni][i] * wts[rl];
        atomicAdd(&Out[(size_t)toks[rl] * DIM + n0 + wn + ni * 16 + col], v);
      }
    }
}

extern "C" void kernel_launch(void* const* d_in, const int* in_sizes, int n_in,
                              void* d_out, int out_size, void* d_ws, size_t ws_size,
                              hipStream_t stream) {
  (void)in_sizes; (void)n_in; (void)out_size; (void)ws_size;
  const float* x         = (const float*)d_in[0];
  const float* norm_attn = (const float*)d_in[1];
  const float* wq        = (const float*)d_in[2];
  const float* wkv_a     = (const float*)d_in[3];
  const float* kv_norm_w = (const float*)d_in[4];
  const float* wkv_b     = (const float*)d_in[5];
  const float* wo        = (const float*)d_in[6];
  const float* norm_moe  = (const float*)d_in[7];
  const float* gate_w    = (const float*)d_in[8];
  const float* gate_b    = (const float*)d_in[9];
  const float* ew1       = (const float*)d_in[10];
  const float* ew2       = (const float*)d_in[11];
  const float* ew3       = (const float*)d_in[12];
  const float* sw1       = (const float*)d_in[13];
  const float* sw2       = (const float*)d_in[14];
  const float* sw3       = (const float*)d_in[15];
  float* out = (float*)d_out;

  char* p = (char*)d_ws;
  auto carve = [&](size_t bytes) -> char* {
    char* r = p;
    p += (bytes + 255) & ~(size_t)255;
    return r;
  };
  u16* xn     = (u16*)carve((size_t)TOK * DIM * 2);
  u16* qbf    = (u16*)carve((size_t)TOK * DIM * 2);
  float* kvf  = (float*)carve((size_t)TOK * (KV_LORA + ROPE) * 4);
  u16* cbf    = (u16*)carve((size_t)TOK * KV_LORA * 2);
  u16* kvpbf  = (u16*)carve((size_t)TOK * 1536 * 2);
  u16* Kb     = (u16*)carve((size_t)16 * 1024 * 128 * 2);
  u16* Vb     = (u16*)carve((size_t)16 * 1024 * 128 * 2);
  u16* attnbf = (u16*)carve((size_t)TOK * DIM * 2);
  u16* xfbf   = (u16*)carve((size_t)TOK * DIM * 2);
  float* xff  = (float*)carve((size_t)TOK * DIM * 4);
  int* ti     = (int*)carve((size_t)TOK * TOPK * 4);
  float* tw   = (float*)carve((size_t)TOK * TOPK * 4);
  int* counts = (int*)carve(N_EXP * 4);
  int* offs   = (int*)carve(N_EXP * 4);
  int* fill   = (int*)carve(N_EXP * 4);
  int* perm   = (int*)carve((size_t)TOK * TOPK * 4);
  float* pwts = (float*)carve((size_t)TOK * TOPK * 4);
  u16* hbuf   = (u16*)carve((size_t)TOK * TOPK * INTER * 2);
  u16* hsh    = (u16*)carve((size_t)TOK * INTER * 2);

  const float qscale = 0.08838834764831845f;  // 128^-0.5

  // MLA
  rmsnorm_kernel<<<TOK, 256, 0, stream>>>(x, norm_attn, xn, nullptr);
  gemm_kernel<1><<<dim3(DIM / 64, TOK / 128), 256, 0, stream>>>(
      xn, wq, nullptr, qbf, nullptr, DIM, DIM, qscale);
  gemm_kernel<0><<<dim3((KV_LORA + ROPE) / 64, TOK / 128), 256, 0, stream>>>(
      xn, wkv_a, kvf, nullptr, nullptr, KV_LORA + ROPE, DIM, 1.0f);
  kvnorm_kernel<<<TOK, 256, 0, stream>>>(kvf, kv_norm_w, cbf);
  gemm_kernel<1><<<dim3(1536 / 64, TOK / 128), 256, 0, stream>>>(
      cbf, wkv_b, nullptr, kvpbf, nullptr, 1536, KV_LORA, 1.0f);
  pack_kv_kernel<<<TOK, 256, 0, stream>>>(kvpbf, kvf, Kb, Vb);
  attn_kernel<<<dim3(1024 / 128, 2 * HEADS), 256, 0, stream>>>(qbf, Kb, Vb, attnbf);
  gemm_kernel<2><<<dim3(DIM / 64, TOK / 128), 256, 0, stream>>>(
      attnbf, wo, out, nullptr, x, DIM, DIM, 1.0f);  // out = x + attn

  // MoE
  rmsnorm_kernel<<<TOK, 256, 0, stream>>>(out, norm_moe, xfbf, xff);
  zero32_kernel<<<1, 32, 0, stream>>>(counts);
  gate_kernel<<<TOK, 32, 0, stream>>>(xff, gate_w, gate_b, ti, tw, counts);
  offsets_kernel<<<1, 32, 0, stream>>>(counts, offs, fill);
  scatter_kernel<<<TOK, 32, 0, stream>>>(ti, tw, offs, fill, perm, pwts);
  // routed experts
  moe_gateup_kernel<<<dim3(INTER / 64, TOK / 64, N_EXP), 256, 0, stream>>>(
      xfbf, ew1, ew3, (long)DIM * INTER, perm, counts, offs, 0, hbuf);
  moe_down_kernel<<<dim3(DIM / 64, TOK / 64, N_EXP), 256, 0, stream>>>(
      hbuf, ew2, (long)INTER * DIM, perm, pwts, counts, offs, 0, out);
  // shared expert
  moe_gateup_kernel<<<dim3(INTER / 64, TOK / 64, 1), 256, 0, stream>>>(
      xfbf, sw1, sw3, 0, nullptr, nullptr, nullptr, TOK, hsh);
  moe_down_kernel<<<dim3(DIM / 64, TOK / 64, 1), 256, 0, stream>>>(
      hsh, sw2, 0, nullptr, nullptr, nullptr, nullptr, TOK, out);
}
